// GuidedAttention_44590350467348
// MI455X (gfx1250) — compile-verified, matrix operand order checked by reading, not run on hardware
//
#include <hip/hip_runtime.h>

#define BQ 64
#define LSEQ 512
#define EDIM 64
#define NHEAD 8
#define HDIM 8
#define NROW (BQ * LSEQ) /* 32768 */

typedef __attribute__((ext_vector_type(16))) _Float16 v16h;
typedef __attribute__((ext_vector_type(8))) float v8f;

#if defined(__has_builtin)
#if __has_builtin(__builtin_amdgcn_global_load_async_to_lds_b128)
#define HAVE_ASYNC_LDS 1
#endif
#endif

// payload type demanded by the async-to-LDS builtin (per hipcc diagnostic:
// param0 is "vector_size(16) int __device__ *", i.e. int4 in AS1)
typedef int v4i_ __attribute__((vector_size(16)));
typedef __attribute__((address_space(1))) v4i_* as1_v4i;
typedef __attribute__((address_space(3))) v4i_* as3_v4i;

// K index held by v16h slot s for a lane in K-half `khalf` (ISA 7.12.2,
// 16-bit A 16x32 layout; B uses the same per-lane K ordering with N=lane%16).
__device__ __forceinline__ int kslot(int s, int khalf) {
  int j = s >> 1, p = s & 1;
  return (j < 4) ? (khalf * 8 + j * 2 + p) : (16 + khalf * 8 + (j - 4) * 2 + p);
}

__device__ __forceinline__ v8f wmma16(v16h a, v16h b, v8f c) {
  return __builtin_amdgcn_wmma_f32_16x16x32_f16(false, a, false, b, (short)0, c,
                                                false, false);
}

// butterfly reductions within each 16-lane half of the wave32 (matches the
// WMMA C-layout row split: lanes 0-15 hold rows 0-7, lanes 16-31 rows 8-15)
__device__ __forceinline__ float hmax16(float v) {
#pragma unroll
  for (int m = 1; m < 16; m <<= 1) v = fmaxf(v, __shfl_xor(v, m, 32));
  return v;
}
__device__ __forceinline__ float hsum16(float v) {
#pragma unroll
  for (int m = 1; m < 16; m <<= 1) v += __shfl_xor(v, m, 32);
  return v;
}

// wave-local LDS fence: CDNA5 split dep-counter wait (s_wait_dscnt)
__device__ __forceinline__ void lds_wave_fence() {
  __builtin_amdgcn_wave_barrier();
  asm volatile("s_wait_dscnt 0x0" ::: "memory");
  __builtin_amdgcn_wave_barrier();
}

// 16-byte global -> LDS copy: async (ASYNCcnt-tracked) when available.
__device__ __forceinline__ void async_cp16(const _Float16* g, _Float16* l) {
#ifdef HAVE_ASYNC_LDS
  __builtin_amdgcn_global_load_async_to_lds_b128((as1_v4i)(void*)g,
                                                 (as3_v4i)(void*)l, 0, 0);
#else
  *(float4*)l = *(const float4*)g;
#endif
}

__device__ __forceinline__ void wait_async() {
#ifdef HAVE_ASYNC_LDS
#if __has_builtin(__builtin_amdgcn_s_wait_asynccnt)
  __builtin_amdgcn_s_wait_asynccnt(0);
#else
  asm volatile("s_wait_asynccnt 0x0" ::: "memory");
#endif
#endif
}

// ---------------------------------------------------------------------------
// Kernel 1: fused QKV projections.  grid=(NROW/16, 3), block=128 (4 waves,
// each wave one 16-col output tile).  out = X @ W^T + b, q pre-scaled 1/sqrt(D)
// ---------------------------------------------------------------------------
__global__ void __launch_bounds__(128)
ga_qkv(const float* __restrict__ q, const float* __restrict__ k,
       const float* __restrict__ Wq, const float* __restrict__ bq,
       const float* __restrict__ Wk, const float* __restrict__ bk,
       const float* __restrict__ Wv, const float* __restrict__ bv,
       _Float16* __restrict__ qh, _Float16* __restrict__ kh,
       _Float16* __restrict__ vh) {
  const int lane = threadIdx.x & 31;
  const int wv = threadIdx.x >> 5;
  const int row0 = blockIdx.x * 16;
  const int which = blockIdx.y;
  const float* X;
  const float* W;
  const float* bias;
  _Float16* out;
  float osc;
  if (which == 0) { X = q; W = Wq; bias = bq; out = qh; osc = 0.35355339059327373f; }
  else if (which == 1) { X = k; W = Wk; bias = bk; out = kh; osc = 1.0f; }
  else { X = k; W = Wv; bias = bv; out = vh; osc = 1.0f; }

  const int m = lane & 15, half = lane >> 4;
  const int n = wv * 16 + m;
  __builtin_prefetch(&W[(size_t)n * EDIM], 0, 3);  // -> global_prefetch_b8
  v8f c;
  const float bn = bias[n];
#pragma unroll
  for (int r = 0; r < 8; ++r) c[r] = bn;

#pragma unroll
  for (int kb = 0; kb < EDIM; kb += 32) {
    v16h a, b;
#pragma unroll
    for (int s = 0; s < 16; ++s) {
      int kk = kb + kslot(s, half);
      a[s] = (_Float16)X[(size_t)(row0 + m) * EDIM + kk];
      b[s] = (_Float16)W[(size_t)n * EDIM + kk];
    }
    c = wmma16(a, b, c);
  }
#pragma unroll
  for (int r = 0; r < 8; ++r)
    out[(size_t)(row0 + r + 8 * half) * EDIM + n] = (_Float16)(c[r] * osc);
}

// ---------------------------------------------------------------------------
// Kernel 2: fused flash attention.  grid=(LSEQ/16, B), block=256 (8 waves,
// wave h = head h, block owns a 16-row Lq tile).  K/V 16x64 tiles are
// double-buffered in LDS via async global->LDS copies (ASYNCcnt) so the
// softmax+WMMA work overlaps the next tile's fill.  Pass 1: online softmax +
// P@V via WMMA.  Pass 2: recompute scores (cheap) to emit head-mean attn_wts
// via LDS ds_add atomics; scores never touch HBM (saves ~512 MB of traffic).
// ---------------------------------------------------------------------------
__global__ void __launch_bounds__(256)
ga_attn(const _Float16* __restrict__ qh, const _Float16* __restrict__ kh,
        const _Float16* __restrict__ vh, _Float16* __restrict__ ctx,
        float* __restrict__ attw) {
  __shared__ float s_attn[16 * LSEQ];        // 32 KB head-mean accumulator
  __shared__ _Float16 s_p[NHEAD][16 * 16];   // 4 KB per-wave C->A relayout
  __shared__ _Float16 s_kt[2][16 * EDIM];    // 4 KB K tile, double-buffered
  __shared__ _Float16 s_vt[2][16 * EDIM];    // 4 KB V tile, double-buffered
  const int tid = threadIdx.x;
  const int lane = tid & 31;
  const int h = tid >> 5;
  const int b = blockIdx.y;
  const int row0 = b * LSEQ + blockIdx.x * 16;  // flattened q row
  const int krow0 = b * LSEQ;
  const int m = lane & 15, half = lane >> 4;

  for (int i = tid; i < 16 * LSEQ; i += 256) s_attn[i] = 0.0f;

  // per-thread 16B chunk of the K/V tile staging (threads 0-127: K, 128-255: V)
  const int chunk = tid & 127;
  const int crow = chunk >> 3, cseg = (chunk & 7) * 8;
  const _Float16* gkv = (tid < 128) ? kh : vh;
  _Float16* ld0 = ((tid < 128) ? s_kt[0] : s_vt[0]) + crow * EDIM + cseg;
  _Float16* ld1 = ((tid < 128) ? s_kt[1] : s_vt[1]) + crow * EDIM + cseg;
  auto stage = [&](int lk0, int buf) {
    async_cp16(gkv + (size_t)(krow0 + lk0 + crow) * EDIM + cseg,
               buf ? ld1 : ld0);
  };

  // A = q-head tile (16 x 8, zero-padded K to 32), already scaled
  v16h aq;
#pragma unroll
  for (int s = 0; s < 16; ++s) {
    int kk = kslot(s, half);
    aq[s] = (kk < HDIM) ? qh[(size_t)(row0 + m) * EDIM + h * HDIM + kk]
                        : (_Float16)0.0f;
  }

  float mrow[8], lrow[8];
#pragma unroll
  for (int r = 0; r < 8; ++r) { mrow[r] = -1e30f; lrow[r] = 0.0f; }
  v8f cacc = {};

  stage(0, 0);
  for (int it = 0; it < LSEQ / 16; ++it) {
    const int buf = it & 1;
    wait_async();
    __syncthreads();  // tile `it` resident in s_kt/s_vt[buf] for all waves
    if (it + 1 < LSEQ / 16) stage((it + 1) * 16, buf ^ 1);  // overlap fill

    // B[d, j] = K[lk0 + j, h*8 + d]  (zero-padded K dim)
    v16h bk16;
#pragma unroll
    for (int s = 0; s < 16; ++s) {
      int kk = kslot(s, half);
      bk16[s] = (kk < HDIM) ? s_kt[buf][m * EDIM + h * HDIM + kk]
                            : (_Float16)0.0f;
    }
    v8f cs = {};
    cs = wmma16(aq, bk16, cs);

    float p[8];
#pragma unroll
    for (int r = 0; r < 8; ++r) {
      float tmax = hmax16(cs[r]);
      float nm = fmaxf(mrow[r], tmax);
      float sc = __expf(mrow[r] - nm);
      lrow[r] *= sc;
      cacc[r] *= sc;
      mrow[r] = nm;
      p[r] = __expf(cs[r] - nm);
      lrow[r] += hsum16(p[r]);
      s_p[h][(r + 8 * half) * 16 + m] = (_Float16)p[r];  // C-layout -> row-major
    }
    lds_wave_fence();

    // A2 = P tile (16 rows x 16 lk, K padded to 32), B2 = V tile (16 x 8)
    v16h ap, bv16;
#pragma unroll
    for (int s = 0; s < 16; ++s) {
      int kk = kslot(s, half);
      ap[s] = (kk < 16) ? s_p[h][m * 16 + kk] : (_Float16)0.0f;
      bv16[s] = (kk < 16 && m < HDIM)
                    ? s_vt[buf][kk * EDIM + h * HDIM + m]
                    : (_Float16)0.0f;
    }
    cacc = wmma16(ap, bv16, cacc);
    lds_wave_fence();
  }

  // normalized context out (f16 staging for the Wo GEMM)
#pragma unroll
  for (int r = 0; r < 8; ++r) {
    float inv = 1.0f / lrow[r];
    if (m < HDIM)
      ctx[(size_t)(row0 + r + 8 * half) * EDIM + h * HDIM + m] =
          (_Float16)(cacc[r] * inv);
  }

  // Pass 2: head-mean attention weights (same async K-tile pipeline)
  __syncthreads();
  stage(0, 0);
  for (int it = 0; it < LSEQ / 16; ++it) {
    const int buf = it & 1;
    wait_async();
    __syncthreads();
    if (it + 1 < LSEQ / 16) stage((it + 1) * 16, buf ^ 1);

    v16h bk16;
#pragma unroll
    for (int s = 0; s < 16; ++s) {
      int kk = kslot(s, half);
      bk16[s] = (kk < HDIM) ? s_kt[buf][m * EDIM + h * HDIM + kk]
                            : (_Float16)0.0f;
    }
    v8f cs = {};
    cs = wmma16(aq, bk16, cs);
#pragma unroll
    for (int r = 0; r < 8; ++r) {
      float pv = __expf(cs[r] - mrow[r]) * (0.125f / lrow[r]);
      atomicAdd(&s_attn[(r + 8 * half) * LSEQ + it * 16 + m], pv);
    }
  }
  __syncthreads();

  const size_t abase =
      (size_t)b * LSEQ * LSEQ + (size_t)(blockIdx.x * 16) * LSEQ;
  for (int i = tid; i < 16 * LSEQ; i += 256) attw[abase + i] = s_attn[i];
}

// ---------------------------------------------------------------------------
// Kernel 3/5: y = LN(A16 @ W^T + bias + resid).  One wave owns a full 16x64
// row tile (4 col tiles) so LN row stats reduce with wave shuffles.
// grid = NROW/128, block = 256 (8 waves = 8 row tiles).
// ---------------------------------------------------------------------------
__global__ void __launch_bounds__(256)
ga_proj_res_ln(const _Float16* __restrict__ A16, const float* __restrict__ W,
               const float* __restrict__ bias, const float* __restrict__ resid,
               const float* __restrict__ g, const float* __restrict__ bb,
               float* __restrict__ xout, _Float16* __restrict__ xout16) {
  const int lane = threadIdx.x & 31;
  const int wv = threadIdx.x >> 5;
  const int row0 = (blockIdx.x * 8 + wv) * 16;
  const int m = lane & 15, half = lane >> 4;
  __builtin_prefetch(&W[(size_t)m * EDIM], 0, 3);

  v8f c[4];
#pragma unroll
  for (int t = 0; t < 4; ++t) {
    float bn = bias[t * 16 + m];
#pragma unroll
    for (int r = 0; r < 8; ++r) c[t][r] = bn;
  }

#pragma unroll
  for (int kb = 0; kb < EDIM; kb += 32) {
    v16h a;
#pragma unroll
    for (int s = 0; s < 16; ++s)
      a[s] = A16[(size_t)(row0 + m) * EDIM + kb + kslot(s, half)];
#pragma unroll
    for (int t = 0; t < 4; ++t) {
      v16h bmat;
#pragma unroll
      for (int s = 0; s < 16; ++s)
        bmat[s] = (_Float16)W[(size_t)(t * 16 + m) * EDIM + kb + kslot(s, half)];
      c[t] = wmma16(a, bmat, c[t]);
    }
  }

#pragma unroll
  for (int t = 0; t < 4; ++t)
#pragma unroll
    for (int r = 0; r < 8; ++r)
      c[t][r] += resid[(size_t)(row0 + r + 8 * half) * EDIM + t * 16 + m];

  float mu[8], rs[8];
#pragma unroll
  for (int r = 0; r < 8; ++r) {
    float sm = 0.f, sq = 0.f;
#pragma unroll
    for (int t = 0; t < 4; ++t) {
      sm += c[t][r];
      sq += c[t][r] * c[t][r];
    }
    sm = hsum16(sm);
    sq = hsum16(sq);
    float mean = sm * (1.0f / 64.0f);
    float var = sq * (1.0f / 64.0f) - mean * mean;
    mu[r] = mean;
    rs[r] = __frsqrt_rn(var + 1e-5f);
  }

#pragma unroll
  for (int t = 0; t < 4; ++t) {
    int col = t * 16 + m;
    float gg = g[col], bv = bb[col];
#pragma unroll
    for (int r = 0; r < 8; ++r) {
      float y = (c[t][r] - mu[r]) * rs[r] * gg + bv;
      size_t idx = (size_t)(row0 + r + 8 * half) * EDIM + col;
      xout[idx] = y;
      if (xout16) xout16[idx] = (_Float16)y;
    }
  }
}

// ---------------------------------------------------------------------------
// Kernel 4: h1 = relu(x @ W1^T + b1) -> f16.  grid=NROW/16, block=128.
// ---------------------------------------------------------------------------
__global__ void __launch_bounds__(128)
ga_ffn1(const _Float16* __restrict__ x16, const float* __restrict__ W1,
        const float* __restrict__ b1, _Float16* __restrict__ h1) {
  const int lane = threadIdx.x & 31;
  const int wv = threadIdx.x >> 5;
  const int row0 = blockIdx.x * 16;
  const int m = lane & 15, half = lane >> 4;
  const int n = wv * 16 + m;
  __builtin_prefetch(&W1[(size_t)n * EDIM], 0, 3);
  v8f c;
  const float bn = b1[n];
#pragma unroll
  for (int r = 0; r < 8; ++r) c[r] = bn;
#pragma unroll
  for (int kb = 0; kb < EDIM; kb += 32) {
    v16h a, b;
#pragma unroll
    for (int s = 0; s < 16; ++s) {
      int kk = kb + kslot(s, half);
      a[s] = x16[(size_t)(row0 + m) * EDIM + kk];
      b[s] = (_Float16)W1[(size_t)n * EDIM + kk];
    }
    c = wmma16(a, b, c);
  }
#pragma unroll
  for (int r = 0; r < 8; ++r)
    h1[(size_t)(row0 + r + 8 * half) * EDIM + n] = (_Float16)fmaxf(c[r], 0.0f);
}

// ---------------------------------------------------------------------------
extern "C" void kernel_launch(void* const* d_in, const int* in_sizes, int n_in,
                              void* d_out, int out_size, void* d_ws,
                              size_t ws_size, hipStream_t stream) {
  const float* q    = (const float*)d_in[0];
  const float* k    = (const float*)d_in[1];
  const float* prev = (const float*)d_in[2];
  const float* Wq   = (const float*)d_in[3];
  const float* bq   = (const float*)d_in[4];
  const float* Wk   = (const float*)d_in[5];
  const float* bk   = (const float*)d_in[6];
  const float* Wv   = (const float*)d_in[7];
  const float* bv   = (const float*)d_in[8];
  const float* Wo   = (const float*)d_in[9];
  const float* bo   = (const float*)d_in[10];
  const float* g1   = (const float*)d_in[11];
  const float* bb1  = (const float*)d_in[12];
  const float* W1   = (const float*)d_in[13];
  const float* b1   = (const float*)d_in[14];
  const float* W2   = (const float*)d_in[15];
  const float* b2   = (const float*)d_in[16];
  const float* g2   = (const float*)d_in[17];
  const float* bb2  = (const float*)d_in[18];

  float* out = (float*)d_out;                          // [B, Lq, E]
  float* attw = out + (size_t)BQ * LSEQ * EDIM;        // [B, Lq, Lk]

  char* ws = (char*)d_ws;                              // 32 MB used
  _Float16* qh  = (_Float16*)(ws + (size_t)0);
  _Float16* kh  = (_Float16*)(ws + ((size_t)4 << 20));
  _Float16* vh  = (_Float16*)(ws + ((size_t)8 << 20));
  _Float16* ctx = (_Float16*)(ws + ((size_t)12 << 20));
  float*    xf  = (float*)   (ws + ((size_t)16 << 20));
  _Float16* x16 = (_Float16*)(ws + ((size_t)24 << 20));
  _Float16* h1  = (_Float16*)(ws + ((size_t)28 << 20));

  ga_qkv<<<dim3(NROW / 16, 3), 128, 0, stream>>>(q, k, Wq, bq, Wk, bk, Wv, bv,
                                                 qh, kh, vh);
  ga_attn<<<dim3(LSEQ / 16, BQ), 256, 0, stream>>>(qh, kh, vh, ctx, attw);
  ga_proj_res_ln<<<NROW / 128, 256, 0, stream>>>(ctx, Wo, bo, prev, g1, bb1,
                                                 xf, x16);
  ga_ffn1<<<NROW / 16, 128, 0, stream>>>(x16, W1, b1, h1);
  ga_proj_res_ln<<<NROW / 128, 256, 0, stream>>>(h1, W2, b2, xf, g2, bb2, out,
                                                 (_Float16*)nullptr);
}
